// Decoder_55293408968757
// MI455X (gfx1250) — matrix operand order repaired and checked by reading, not compiled, not run on hardware
//
#include <hip/hip_runtime.h>
#include <hip/hip_bf16.h>

// ---------------------------------------------------------------------------
// Tacotron2-style decoder for MI455X (gfx1250, wave32, WMMA).
// Persistent-kernel design: weights converted once to bf16 WMMA fragments
// (L2-resident, 192MB L2 >> 38MB of bf16 weights), 400 recurrent steps run
// inside one kernel with a global-memory grid barrier between phases.
// All big matmuls use v_wmma_f32_16x16x32_bf16 (fp32 accumulation).
// ---------------------------------------------------------------------------

#define NBLK 64
#define NTHR 256
#define NT_ALL (NBLK * NTHR)

typedef __attribute__((ext_vector_type(4)))  unsigned int u32x4;
typedef __attribute__((ext_vector_type(16))) __bf16       bf16x16;
typedef __attribute__((ext_vector_type(8)))  float        f32x8;

union FragBF { bf16x16 v; u32x4 q[2]; };

// ---- workspace layout (bytes) ---------------------------------------------
constexpr size_t OFF_BAR  = 0;                 // 2 x u32 barrier, padded
constexpr size_t SZ_BAR   = 256;
constexpr size_t OFF_WA   = OFF_BAR + SZ_BAR;  // att gates W: K=1792,N=4096 -> 56x256 frags
constexpr size_t SZ_WA    = (size_t)56 * 256 * 512 * 2;
constexpr size_t OFF_WD   = OFF_WA + SZ_WA;    // dec gates W: K=2560,N=4096 -> 80x256 frags
constexpr size_t SZ_WD    = (size_t)80 * 256 * 512 * 2;
constexpr size_t OFF_WQ   = OFF_WD + SZ_WD;    // Wq: K=1024,N=1024 -> 32x64 frags
constexpr size_t SZ_WQ    = (size_t)32 * 64 * 512 * 2;
constexpr size_t OFF_WL   = OFF_WQ + SZ_WQ;    // loc_w: K=32,N=1024 -> 1x64 frags
constexpr size_t SZ_WL    = (size_t)1 * 64 * 512 * 2;
constexpr size_t OFF_PRE  = OFF_WL + SZ_WL;    // prenet out bf16 [400][32][256]
constexpr size_t SZ_PRE   = (size_t)400 * 32 * 256 * 2;
constexpr size_t OFF_WENC = OFF_PRE + SZ_PRE;  // W_enc f32 [32*200][1024]
constexpr size_t SZ_WENC  = (size_t)6400 * 1024 * 4;
constexpr size_t OFF_H1   = OFF_WENC + SZ_WENC; // prenet hidden f32 [12800][256]
constexpr size_t SZ_H1    = (size_t)12800 * 256 * 4;
constexpr size_t OFF_ATTC = OFF_H1 + SZ_H1;     // f32 [32][1024]
constexpr size_t OFF_DECC = OFF_ATTC + 131072;  // f32 [32][1024]
constexpr size_t OFF_DECH = OFF_DECC + 131072;  // f32 [32][1024]
constexpr size_t OFF_CTX  = OFF_DECH + 131072;  // f32 [32][512]
constexpr size_t OFF_W    = OFF_CTX + 65536;    // f32 [32][256] (Te=200 padded)
constexpr size_t OFF_WCUM = OFF_W + 32768;      // f32 [32][256]
constexpr size_t OFF_E    = OFF_WCUM + 32768;   // f32 [32][256]
constexpr size_t OFF_Q    = OFF_E + 32768;      // f32 [32][1024]
constexpr size_t OFF_ZA   = OFF_Q + 131072;     // f32 [32][4096]
constexpr size_t OFF_ZD   = OFF_ZA + 524288;    // f32 [32][4096]
constexpr size_t OFF_XA   = OFF_ZD + 524288;    // bf16 [32][1792] = [pre|ctx|att_h]
constexpr size_t OFF_XD   = OFF_XA + 114688;    // bf16 [32][2560] = [att_h|ctx|dec_h]
constexpr size_t OFF_LOC  = OFF_XD + 163840;    // bf16 [6400][32] conv features
constexpr size_t WS_NEED  = OFF_LOC + 409600;

// output regions (floats)
constexpr size_t OUT_MEL   = 0;                 // [32][400][80]
constexpr size_t OUT_GATE  = 1024000;           // [400*32] time-major
constexpr size_t OUT_ALIGN = 1036800;           // [32][400][200]

// ---------------------------------------------------------------------------
__device__ __forceinline__ unsigned short f2bf(float x) {
  union { float f; unsigned u; } v; v.f = x;
  unsigned r = v.u + 0x7FFFu + ((v.u >> 16) & 1u);   // round-to-nearest-even
  return (unsigned short)(r >> 16);
}
__device__ __forceinline__ float sigm(float x) { return 1.0f / (1.0f + __expf(-x)); }

__device__ __forceinline__ void grid_sync(unsigned* bar) {
  __syncthreads();
  if (threadIdx.x == 0) {
    __threadfence();
    unsigned ticket = __hip_atomic_fetch_add(&bar[0], 1u, __ATOMIC_ACQ_REL,
                                             __HIP_MEMORY_SCOPE_AGENT);
    unsigned gen = ticket / (unsigned)NBLK;
    if (ticket % (unsigned)NBLK == (unsigned)NBLK - 1u) {
      __hip_atomic_store(&bar[1], gen + 1u, __ATOMIC_RELEASE, __HIP_MEMORY_SCOPE_AGENT);
    } else {
      while (__hip_atomic_load(&bar[1], __ATOMIC_ACQUIRE, __HIP_MEMORY_SCOPE_AGENT) <= gen) {
        __builtin_amdgcn_s_sleep(2);
      }
    }
    __threadfence();
  }
  __syncthreads();
}

// C = X(32 x K, bf16 row-major, leading dim ldx) @ packed-W ; one 16x16 tile
// per wave; tile = mt + 2*nt. A-frag layout per ISA 7.12.2 (16-bit A 16x32).
__device__ __forceinline__ void wmma_gemm_rowmajor(
    const unsigned short* __restrict__ X, int ldx,
    const unsigned short* __restrict__ PW, int KT,
    float* __restrict__ Z, int ldz, int tile, int lane) {
  int mt = tile & 1, nt = tile >> 1;
  const unsigned short* ap = X + (size_t)(mt * 16 + (lane & 15)) * ldx + ((lane >> 4) << 3);
  const unsigned short* bp = PW + (size_t)(nt * KT) * 512 + lane * 16;
  f32x8 c = {0.f,0.f,0.f,0.f,0.f,0.f,0.f,0.f};
  for (int kt = 0; kt < KT; ++kt) {
    FragBF a, b;
    a.q[0] = *(const u32x4*)(ap);
    a.q[1] = *(const u32x4*)(ap + 16);   // K jumps by 16 within a lane
    b.q[0] = *(const u32x4*)(bp);
    b.q[1] = *(const u32x4*)(bp + 8);    // packed: 16 contiguous halves/lane
    __builtin_prefetch(bp + 512, 0, 1);
    c = __builtin_amdgcn_wmma_f32_16x16x32_bf16(false, a.v, false, b.v,
                                                (short)0, c, false, false);
    ap += 32; bp += 512;
  }
  int row0 = mt * 16 + ((lane >> 4) << 3);
  int col  = nt * 16 + (lane & 15);
#pragma unroll
  for (int r = 0; r < 8; ++r) Z[(size_t)(row0 + r) * ldz + col] = c[r];
}

__device__ __forceinline__ void lstm_pointwise(
    const float* __restrict__ z, const float* __restrict__ bias,
    float* __restrict__ cst, float* __restrict__ hdst,
    unsigned short* __restrict__ bfA, int ldA, int offA,
    unsigned short* __restrict__ bfB, int ldB, int offB, int g) {
  for (int idx = g; idx < 32 * 1024; idx += NT_ALL) {
    int b = idx >> 10, n = idx & 1023;
    const float* zr = z + (size_t)b * 4096;
    float i_ = zr[n]          + bias[n];
    float f_ = zr[n + 1024]   + bias[n + 1024];
    float g_ = zr[n + 2048]   + bias[n + 2048];
    float o_ = zr[n + 3072]   + bias[n + 3072];
    float c  = sigm(f_) * cst[idx] + sigm(i_) * tanhf(g_);
    float h  = sigm(o_) * tanhf(c);
    cst[idx] = c;
    if (hdst) hdst[idx] = h;
    unsigned short hb = f2bf(h);
    if (bfA) bfA[(size_t)b * ldA + offA + n] = hb;
    if (bfB) bfB[(size_t)b * ldB + offB + n] = hb;
  }
}

// ---------------------------------------------------------------------------
// one-time kernels
// ---------------------------------------------------------------------------
// Pack W (virtually [W0;W1], K rows x N cols, f32) into WMMA B fragments:
// frag = nt*KT + kt ; lane l, half j -> W[kt*32 + (l>=16)*8 + j + (j>=8)*8][nt*16 + l%16]
__global__ void pack_w2(const float* __restrict__ W0, int K0,
                        const float* __restrict__ W1,
                        unsigned short* __restrict__ dst, int K, int N) {
  int g = blockIdx.x * blockDim.x + threadIdx.x;
  int KT = K >> 5, NT = N >> 4;
  int total = KT * NT * 32;
  if (g >= total) return;
  int lane = g & 31, frag = g >> 5;
  int kt = frag % KT, nt = frag / KT;
  int col = nt * 16 + (lane & 15);
  int kb  = kt * 32 + ((lane >> 4) << 3);
  unsigned short* o = dst + (size_t)frag * 512 + lane * 16;
#pragma unroll
  for (int j = 0; j < 16; ++j) {
    int k = kb + j + ((j >> 3) << 3);
    float v = (k < K0) ? W0[(size_t)k * N + col] : W1[(size_t)(k - K0) * N + col];
    o[j] = f2bf(v);
  }
}

__global__ void prenet1_k(const float* __restrict__ mel, const float* __restrict__ w1,
                          const float* __restrict__ b1, float* __restrict__ h1) {
  int g = blockIdx.x * blockDim.x + threadIdx.x;
  if (g >= 400 * 32 * 256) return;
  int j = g & 255; int tb = g >> 8; int b = tb & 31; int t = tb >> 5;
  float s = b1[j];
  if (t > 0) {
    const float* x = mel + ((size_t)b * 400 + (t - 1)) * 80;   // teacher-forced shift
    for (int k = 0; k < 80; ++k) s += x[k] * w1[k * 256 + j];
  }
  h1[(size_t)g] = fmaxf(s, 0.f);
}

__global__ void prenet2_k(const float* __restrict__ h1, const float* __restrict__ w2,
                          const float* __restrict__ b2, unsigned short* __restrict__ pre) {
  int g = blockIdx.x * blockDim.x + threadIdx.x;
  if (g >= 400 * 32 * 256) return;
  int j = g & 255; int tb = g >> 8;
  const float* x = h1 + (size_t)tb * 256;
  float s = b2[j];
  for (int k = 0; k < 256; ++k) s += x[k] * w2[k * 256 + j];
  pre[(size_t)g] = f2bf(fmaxf(s, 0.f));
}

__global__ void wenc_k(const float* __restrict__ enc, const float* __restrict__ Wm,
                       float* __restrict__ We) {
  int g = blockIdx.x * blockDim.x + threadIdx.x;
  if (g >= 6400 * 1024) return;
  int n = g & 1023; int bt = g >> 10;
  const float* x = enc + (size_t)bt * 512;
  float s = 0.f;
  for (int k = 0; k < 512; ++k) s += x[k] * Wm[k * 1024 + n];
  We[(size_t)g] = s;
}

// ---------------------------------------------------------------------------
// persistent decoder kernel
// ---------------------------------------------------------------------------
__launch_bounds__(NTHR)
__global__ void decoder_persistent(
    const float* __restrict__ enc_out, const unsigned char* __restrict__ mask,
    const float* __restrict__ att_b, const float* __restrict__ loc_conv,
    const float* __restrict__ v_w, const float* __restrict__ dec_b,
    const float* __restrict__ proj_w, const float* __restrict__ proj_b,
    const float* __restrict__ gate_w, const float* __restrict__ gate_b,
    unsigned char* __restrict__ ws, float* __restrict__ out) {
  unsigned*       bar    = (unsigned*)(ws + OFF_BAR);
  const unsigned short* pWa = (const unsigned short*)(ws + OFF_WA);
  const unsigned short* pWd = (const unsigned short*)(ws + OFF_WD);
  const unsigned short* pWq = (const unsigned short*)(ws + OFF_WQ);
  const unsigned short* pWl = (const unsigned short*)(ws + OFF_WL);
  const unsigned short* pre = (const unsigned short*)(ws + OFF_PRE);
  const float*    Wenc   = (const float*)(ws + OFF_WENC);
  float* att_c = (float*)(ws + OFF_ATTC);
  float* dec_c = (float*)(ws + OFF_DECC);
  float* dec_h = (float*)(ws + OFF_DECH);
  float* ctx   = (float*)(ws + OFF_CTX);
  float* wbuf  = (float*)(ws + OFF_W);
  float* wcum  = (float*)(ws + OFF_WCUM);
  float* ebuf  = (float*)(ws + OFF_E);
  float* qbuf  = (float*)(ws + OFF_Q);
  float* z_att = (float*)(ws + OFF_ZA);
  float* z_dec = (float*)(ws + OFF_ZD);
  unsigned short* xa  = (unsigned short*)(ws + OFF_XA);
  unsigned short* xd  = (unsigned short*)(ws + OFF_XD);
  unsigned short* locb = (unsigned short*)(ws + OFF_LOC);

  const int g    = blockIdx.x * NTHR + threadIdx.x;
  const int wave = g >> 5;
  const int lane = threadIdx.x & 31;

  // ---- init: zero recurrent state, stage xa = [pre_0 | 0 | 0] -------------
  for (int i = g; i < 32 * 1024; i += NT_ALL) { att_c[i] = 0.f; dec_c[i] = 0.f; dec_h[i] = 0.f; }
  for (int i = g; i < 32 * 256;  i += NT_ALL) { wbuf[i] = 0.f; wcum[i] = 0.f; }
  for (int i = g; i < 32 * 512;  i += NT_ALL) ctx[i] = 0.f;
  for (int i = g; i < 32 * 1792; i += NT_ALL) {
    int b = i / 1792, j = i - b * 1792;
    xa[i] = (j < 256) ? pre[(size_t)b * 256 + j] : (unsigned short)0;
  }
  for (int i = g; i < 32 * 2560; i += NT_ALL) xd[i] = 0;
  grid_sync(bar);

  for (int t = 0; t < 400; ++t) {
    // P0: att-LSTM gates z = [pre|ctx|h_att] @ [att_k;att_r]  (512 tiles)
    wmma_gemm_rowmajor(xa, 1792, pWa, 56, z_att, 4096, wave, lane);
    grid_sync(bar);

    // P1: att-LSTM pointwise -> att_h (bf16 into xa[768..], xd[0..])
    lstm_pointwise(z_att, att_b, att_c, nullptr, xa, 1792, 768, xd, 2560, 0, g);
    grid_sync(bar);

    // P2: q = att_h @ Wq (blocks 0..15, 128 tiles)  ||  location conv (rest)
    if (blockIdx.x < 16) {
      wmma_gemm_rowmajor(xd, 2560, pWq, 32, qbuf, 1024, wave, lane);
    } else {
      int tid = (blockIdx.x - 16) * NTHR + threadIdx.x;
      for (int idx = tid; idx < 32 * 200 * 32; idx += 48 * NTHR) {
        int f = idx & 31; int rem = idx >> 5;
        int b = rem / 200; int tt = rem - b * 200;
        float s = 0.f;
        for (int dk = 0; dk < 31; ++dk) {
          int ts = tt + dk - 15;                       // SAME padding
          if (ts >= 0 && ts < 200) {
            s += wbuf[b * 256 + ts] * loc_conv[(dk * 2 + 0) * 32 + f]
               + wcum[b * 256 + ts] * loc_conv[(dk * 2 + 1) * 32 + f];
          }
        }
        locb[(size_t)rem * 32 + f] = f2bf(s);
      }
    }
    grid_sync(bar);

    // P3: energies e[b][t'] = v . tanh(q + W_enc + loc@loc_w)  (K=32 -> 1 WMMA)
    if (wave < 400) {
      int mt = wave;
      const unsigned short* ap = locb + (size_t)(mt * 16 + (lane & 15)) * 32 + ((lane >> 4) << 3);
      FragBF a; a.q[0] = *(const u32x4*)ap; a.q[1] = *(const u32x4*)(ap + 16);
      int row0 = mt * 16 + ((lane >> 4) << 3);
      int bArr[8], tArr[8];
#pragma unroll
      for (int r = 0; r < 8; ++r) {
        int row = row0 + r; int b_ = row / 200;
        bArr[r] = b_; tArr[r] = row - b_ * 200;
      }
      float rs[8] = {0.f,0.f,0.f,0.f,0.f,0.f,0.f,0.f};
      const unsigned short* bp = pWl + lane * 16;
      for (int nt = 0; nt < 64; ++nt) {
        FragBF b; b.q[0] = *(const u32x4*)bp; b.q[1] = *(const u32x4*)(bp + 8);
        f32x8 c = {0.f,0.f,0.f,0.f,0.f,0.f,0.f,0.f};
        c = __builtin_amdgcn_wmma_f32_16x16x32_bf16(false, a.v, false, b.v,
                                                    (short)0, c, false, false);
        int col = nt * 16 + (lane & 15);
        float vv = v_w[col];
#pragma unroll
        for (int r = 0; r < 8; ++r) {
          float x = c[r] + qbuf[(size_t)bArr[r] * 1024 + col]
                         + Wenc[(size_t)(row0 + r) * 1024 + col];
          rs[r] += tanhf(x) * vv;
        }
        bp += 512;
      }
#pragma unroll
      for (int r = 0; r < 8; ++r) {       // reduce over the 16 lanes of a half
        float s = rs[r];
        s += __shfl_xor(s, 1, 32); s += __shfl_xor(s, 2, 32);
        s += __shfl_xor(s, 4, 32); s += __shfl_xor(s, 8, 32);
        if ((lane & 15) == 0) {
          float bias = mask[bArr[r] * 200 + tArr[r]] ? 0.f : -1e9f;
          ebuf[bArr[r] * 256 + tArr[r]] = s + bias;
        }
      }
    }
    grid_sync(bar);

    // P4: softmax over Te, update w / w_cum, emit alignments
    if (wave < 32) {
      int b = wave;
      float mx = -3.0e38f;
      for (int k = lane; k < 200; k += 32) mx = fmaxf(mx, ebuf[b * 256 + k]);
      for (int off = 16; off; off >>= 1) mx = fmaxf(mx, __shfl_xor(mx, off, 32));
      float sum = 0.f;
      for (int k = lane; k < 200; k += 32) {
        float ex = __expf(ebuf[b * 256 + k] - mx); ebuf[b * 256 + k] = ex; sum += ex;
      }
      for (int off = 16; off; off >>= 1) sum += __shfl_xor(sum, off, 32);
      float inv = 1.f / sum;
      float* al = out + OUT_ALIGN + ((size_t)b * 400 + t) * 200;
      for (int k = lane; k < 200; k += 32) {
        float wv = ebuf[b * 256 + k] * inv;
        wbuf[b * 256 + k] = wv;
        wcum[b * 256 + k] += wv;
        al[k] = wv;
      }
    }
    grid_sync(bar);

    // P5: context = w @ enc_out ; stage bf16 into xa[256..], xd[1024..]
    {
      int b = g >> 9, ei = g & 511;
      float s = 0.f;
      for (int tt = 0; tt < 200; ++tt)
        s += wbuf[b * 256 + tt] * enc_out[((size_t)b * 200 + tt) * 512 + ei];
      ctx[(size_t)b * 512 + ei] = s;
      unsigned short h = f2bf(s);
      xa[(size_t)b * 1792 + 256 + ei]  = h;
      xd[(size_t)b * 2560 + 1024 + ei] = h;
    }
    grid_sync(bar);

    // P6: dec-LSTM gates z = [h_att|ctx|h_dec] @ [dec_k;dec_r]
    wmma_gemm_rowmajor(xd, 2560, pWd, 80, z_dec, 4096, wave, lane);
    grid_sync(bar);

    // P7: dec-LSTM pointwise -> dec_h (f32 + bf16 into xd[1536..])
    lstm_pointwise(z_dec, dec_b, dec_c, dec_h, xd, 2560, 1536, nullptr, 0, 0, g);
    grid_sync(bar);

    // P8: projections (mel + gate) and stage pre_{t+1} into xa
    if (g < 32 * 81) {
      int b = g / 81, j = g - b * 81;
      const float* dh = dec_h + (size_t)b * 1024;
      const float* cx = ctx + (size_t)b * 512;
      if (j < 80) {
        float s = proj_b[j];
        for (int k = 0; k < 1024; ++k) s += dh[k] * proj_w[(size_t)k * 80 + j];
        for (int k = 0; k < 512;  ++k) s += cx[k] * proj_w[(size_t)(1024 + k) * 80 + j];
        out[OUT_MEL + ((size_t)b * 400 + t) * 80 + j] = s;
      } else {
        float s = gate_b[0];
        for (int k = 0; k < 1024; ++k) s += dh[k] * gate_w[k];
        for (int k = 0; k < 512;  ++k) s += cx[k] * gate_w[1024 + k];
        out[OUT_GATE + (size_t)t * 32 + b] = s;
      }
    }
    if (t + 1 < 400) {
      for (int i = g; i < 32 * 256; i += NT_ALL) {
        int b = i >> 8, j = i & 255;
        xa[(size_t)b * 1792 + j] = pre[((size_t)(t + 1) * 32 + b) * 256 + j];
      }
    }
    grid_sync(bar);
  }
}

// ---------------------------------------------------------------------------
extern "C" void kernel_launch(void* const* d_in, const int* in_sizes, int n_in,
                              void* d_out, int out_size, void* d_ws, size_t ws_size,
                              hipStream_t stream) {
  (void)in_sizes; (void)n_in; (void)out_size; (void)ws_size;
  const float* enc_out  = (const float*)d_in[0];
  const float* mel_gt   = (const float*)d_in[1];
  const unsigned char* mask = (const unsigned char*)d_in[2];
  const float* pw1   = (const float*)d_in[3];
  const float* pb1   = (const float*)d_in[4];
  const float* pw2   = (const float*)d_in[5];
  const float* pb2   = (const float*)d_in[6];
  const float* att_k = (const float*)d_in[7];
  const float* att_r = (const float*)d_in[8];
  const float* att_b = (const float*)d_in[9];
  const float* Wq    = (const float*)d_in[10];
  const float* Wm    = (const float*)d_in[11];
  const float* loc_c = (const float*)d_in[12];
  const float* loc_w = (const float*)d_in[13];
  const float* v_w   = (const float*)d_in[14];
  const float* dec_k = (const float*)d_in[15];
  const float* dec_r = (const float*)d_in[16];
  const float* dec_b = (const float*)d_in[17];
  const float* proj_w = (const float*)d_in[18];
  const float* proj_b = (const float*)d_in[19];
  const float* gate_w = (const float*)d_in[20];
  const float* gate_b = (const float*)d_in[21];
  unsigned char* ws = (unsigned char*)d_ws;
  float* out = (float*)d_out;

  hipMemsetAsync(ws + OFF_BAR, 0, SZ_BAR, stream);

  // weight packing into bf16 WMMA fragments
  { int tot = 56 * 256 * 32;
    pack_w2<<<(tot + 255) / 256, 256, 0, stream>>>(att_k, 768, att_r,
        (unsigned short*)(ws + OFF_WA), 1792, 4096); }
  { int tot = 80 * 256 * 32;
    pack_w2<<<(tot + 255) / 256, 256, 0, stream>>>(dec_k, 1536, dec_r,
        (unsigned short*)(ws + OFF_WD), 2560, 4096); }
  { int tot = 32 * 64 * 32;
    pack_w2<<<(tot + 255) / 256, 256, 0, stream>>>(Wq, 1024, Wq,
        (unsigned short*)(ws + OFF_WQ), 1024, 1024); }
  { int tot = 1 * 64 * 32;
    pack_w2<<<(tot + 255) / 256, 256, 0, stream>>>(loc_w, 32, loc_w,
        (unsigned short*)(ws + OFF_WL), 32, 1024); }

  // prenet over all teacher-forced frames
  { int tot = 400 * 32 * 256;
    prenet1_k<<<(tot + 255) / 256, 256, 0, stream>>>(mel_gt, pw1, pb1,
        (float*)(ws + OFF_H1));
    prenet2_k<<<(tot + 255) / 256, 256, 0, stream>>>((float*)(ws + OFF_H1), pw2, pb2,
        (unsigned short*)(ws + OFF_PRE)); }

  // processed memory W_enc = enc_out @ Wm
  { int tot = 6400 * 1024;
    wenc_k<<<(tot + 255) / 256, 256, 0, stream>>>(enc_out, Wm,
        (float*)(ws + OFF_WENC)); }

  // persistent 400-step decoder
  decoder_persistent<<<NBLK, NTHR, 0, stream>>>(
      enc_out, mask, att_b, loc_c, v_w, dec_b, proj_w, proj_b, gate_w, gate_b,
      ws, out);
}
